// NeuralGraphFingerprint_82197084111091
// MI455X (gfx1250) — matrix operand order repaired and checked by reading, not compiled, not run on hardware
//
#include <hip/hip_runtime.h>
#include <hip/hip_bf16.h>
#include <math.h>

// Problem constants (match reference)
#define NN 100000
#define EE 1600000
#define CC 128
#define FPD 512
#define LL 3
#define GG 1024

typedef __attribute__((ext_vector_type(16))) __bf16 v16bf;
typedef __attribute__((ext_vector_type(8)))  float  v8f;

// Packed-weight sizes (in __bf16 elements), fragment-major:
//   [tile][kchunk][lane][16]  with tile = J/16 tiles, kchunk = 128/32 = 4.
#define WSB_PER_LAYER  ((CC / 16) * 4 * 32 * 16)    // 16384 (Ws or Wn)
#define WFPB_PER_LAYER ((FPD / 16) * 4 * 32 * 16)   // 65536

// ---------------------------------------------------------------------------
// Branch-free tanh: tanh(v) = 1 - 2/(exp(2v)+1).
// Lowers to v_exp_f32 + v_rcp_f32 + FMAs (no EXEC-divergent libm path);
// saturates exactly to +-1 at the extremes (exp->inf -> rcp->0, exp->0 -> -1),
// which matches this workload (W_SCALE=5 => deeply saturated activations).
// ---------------------------------------------------------------------------
__device__ __forceinline__ float fast_tanh(float v) {
    return 1.0f - 2.0f * __builtin_amdgcn_rcpf(__expf(2.0f * v) + 1.0f);
}

// ---------------------------------------------------------------------------
// A-fragment for v_wmma_f32_16x16x32_bf16 from a row-major bf16 tile in LDS.
// Contiguous K order (matches packed B): lanes 0-15 hold row M=lane, elems
// 0..15 = K kc+0..15; lanes 16-31 hold row M=lane-16,
// elems 0..15 = K kc+16..31.
// ---------------------------------------------------------------------------
__device__ __forceinline__ v16bf frag_a_lds(const __bf16* base, int lane, int kc) {
    const int r  = lane & 15;
    const int kb = kc + ((lane >> 4) << 4);
    const __bf16* p = base + r * CC;
    v16bf a;
#pragma unroll
    for (int i = 0; i < 8; ++i) a[i]     = p[kb + i];
#pragma unroll
    for (int i = 0; i < 8; ++i) a[8 + i] = p[kb + 8 + i];
    return a;
}

// ---------------------------------------------------------------------------
// B-fragment from the fragment-major packed bf16 weight array: one aligned
// 32-byte load per lane, no conversions in the hot loop.
// ---------------------------------------------------------------------------
__device__ __forceinline__ v16bf frag_b_packed(const __bf16* __restrict__ P,
                                               int tile, int kchunk, int lane) {
    const size_t idx = ((((size_t)tile * 4 + kchunk) * 32) + lane) << 4;
    return *reinterpret_cast<const v16bf*>(P + idx);
}

// ---------------------------------------------------------------------------
// One-time repack: W[J][128] f32 (row-major, layers contiguous so J_total =
// L*J) -> fragment-major bf16.  B-layout (32x16 K x N): lanes 0-15 hold col
// n=lane, K kc..kc+15; lanes 16-31 hold col n=lane-16, K kc+16..kc+31.
// One thread per (tile, kchunk, lane) -> writes 16 contiguous bf16 (32 B).
// ---------------------------------------------------------------------------
__global__ void pack_weights_kernel(const float* __restrict__ W,
                                    __bf16* __restrict__ P, int total_frags) {
    const int t = blockIdx.x * blockDim.x + threadIdx.x;
    if (t >= total_frags) return;
    const int lane  = t & 31;
    const int chunk = (t >> 5) & 3;
    const int tile  = t >> 7;
    const int n  = lane & 15;
    const int kb = chunk * 32 + ((lane >> 4) << 4);
    const float* p = W + (size_t)(tile * 16 + n) * CC + kb;
    __bf16* q = P + ((size_t)t << 4);
#pragma unroll
    for (int i = 0; i < 16; i += 4) {
        const float4 f = *reinterpret_cast<const float4*>(p + i);
        q[i]     = (__bf16)f.x;
        q[i + 1] = (__bf16)f.y;
        q[i + 2] = (__bf16)f.z;
        q[i + 3] = (__bf16)f.w;
    }
}

// ---------------------------------------------------------------------------
// Zero a buffer (float4 granularity, grid-stride).
// ---------------------------------------------------------------------------
__global__ void zero_f32_kernel(float* __restrict__ p, int n4) {
    float4* q = reinterpret_cast<float4*>(p);
    const float4 z = {0.f, 0.f, 0.f, 0.f};
    for (int i = blockIdx.x * blockDim.x + threadIdx.x; i < n4;
         i += gridDim.x * blockDim.x)
        q[i] = z;
}

// ---------------------------------------------------------------------------
// Edge gather + scatter-add: neigh[row[e]] += x[col[e]].  One wave32 per
// edge; each lane moves 4 contiguous channels (float4 gather, 4 f32 atomics).
// This HBM/atomic-bound stage dominates the roofline (~1.6 GB/layer).
// ---------------------------------------------------------------------------
__global__ __launch_bounds__(256)
void gather_scatter_kernel(const float* __restrict__ x,
                           const int* __restrict__ ei,
                           float* __restrict__ neigh) {
    const int wave = (blockIdx.x * blockDim.x + threadIdx.x) >> 5;
    const int lane = threadIdx.x & 31;
    if (wave >= EE) return;
    const int dst = ei[wave];        // row (scatter target)
    const int src = ei[EE + wave];   // col (gather source)
    const float4 v =
        *reinterpret_cast<const float4*>(x + (size_t)src * CC + lane * 4);
    float* p = neigh + (size_t)dst * CC + lane * 4;
    atomicAdd(p + 0, v.x);
    atomicAdd(p + 1, v.y);
    atomicAdd(p + 2, v.z);
    atomicAdd(p + 3, v.w);
}

// ---------------------------------------------------------------------------
// x_out = tanh(x @ Ws^T + bs + neigh @ Wn^T + bn)
// Block = 16 rows x 128 cols; 8 waves, each owns one 16x16 output tile.
// x/neigh rows staged to LDS as bf16; 8 v_wmma_f32_16x16x32_bf16 per wave,
// B fragments are single 32-B loads from the packed bf16 weights.
// Each block reads and writes only its own 16 rows, so in-place x is safe.
// ---------------------------------------------------------------------------
__global__ __launch_bounds__(256)
void layer_gemm_tanh_kernel(const float* __restrict__ x_src,
                            const float* __restrict__ neigh,
                            const __bf16* __restrict__ Wsb,
                            const float* __restrict__ bs,
                            const __bf16* __restrict__ Wnb,
                            const float* __restrict__ bn,
                            float* __restrict__ x_out) {
    __shared__ __bf16 xs[16][CC];
    __shared__ __bf16 ns[16][CC];

    const int tid = threadIdx.x;
    const int r0  = blockIdx.x * 16;

    // Stage 16x128 of x and neigh, converting f32 -> bf16 (8 elems/thread).
    {
        const int base = tid * 8;
        const int r = base >> 7, c = base & (CC - 1);
        const float4* px =
            reinterpret_cast<const float4*>(x_src + (size_t)r0 * CC + base);
        const float4* pn =
            reinterpret_cast<const float4*>(neigh + (size_t)r0 * CC + base);
        const float4 a0 = px[0], a1 = px[1];
        const float4 b0 = pn[0], b1 = pn[1];
        xs[r][c + 0] = (__bf16)a0.x; xs[r][c + 1] = (__bf16)a0.y;
        xs[r][c + 2] = (__bf16)a0.z; xs[r][c + 3] = (__bf16)a0.w;
        xs[r][c + 4] = (__bf16)a1.x; xs[r][c + 5] = (__bf16)a1.y;
        xs[r][c + 6] = (__bf16)a1.z; xs[r][c + 7] = (__bf16)a1.w;
        ns[r][c + 0] = (__bf16)b0.x; ns[r][c + 1] = (__bf16)b0.y;
        ns[r][c + 2] = (__bf16)b0.z; ns[r][c + 3] = (__bf16)b0.w;
        ns[r][c + 4] = (__bf16)b1.x; ns[r][c + 5] = (__bf16)b1.y;
        ns[r][c + 6] = (__bf16)b1.z; ns[r][c + 7] = (__bf16)b1.w;
    }
    __syncthreads();

    const int w    = tid >> 5;
    const int lane = tid & 31;

    v8f acc = {};
#pragma unroll
    for (int kc = 0; kc < 4; ++kc) {
        const v16bf a = frag_a_lds(&xs[0][0], lane, kc * 32);
        const v16bf b = frag_b_packed(Wsb, w, kc, lane);
        acc = __builtin_amdgcn_wmma_f32_16x16x32_bf16(false, a, false, b,
                                                      (short)0, acc, false, false);
    }
#pragma unroll
    for (int kc = 0; kc < 4; ++kc) {
        const v16bf a = frag_a_lds(&ns[0][0], lane, kc * 32);
        const v16bf b = frag_b_packed(Wnb, w, kc, lane);
        acc = __builtin_amdgcn_wmma_f32_16x16x32_bf16(false, a, false, b,
                                                      (short)0, acc, false, false);
    }

    // C/D layout: lanes 0-15 -> N=lane, M=e; lanes 16-31 -> N=lane-16, M=e+8.
    const int n    = w * 16 + (lane & 15);
    const int mb   = (lane >> 4) << 3;
    const float bb = bs[n] + bn[n];
#pragma unroll
    for (int e = 0; e < 8; ++e)
        x_out[(size_t)(r0 + mb + e) * CC + n] = fast_tanh(acc[e] + bb);
}

// ---------------------------------------------------------------------------
// logits = x @ Wfp^T (16 x 512 per block via WMMA), row softmax, then
// fingerprint[batch[r]] += softmax row (f32 atomics; G=1024 graphs).
// ---------------------------------------------------------------------------
__global__ __launch_bounds__(256)
void fp_softmax_scatter_kernel(const float* __restrict__ x,
                               const __bf16* __restrict__ Wfpb,
                               const int* __restrict__ batch,
                               float* __restrict__ out) {
    __shared__ __bf16 xs[16][CC];
    __shared__ float  ls[16][FPD + 8];  // +8 pad: breaks 64-bank conflicts

    const int tid = threadIdx.x;
    const int r0  = blockIdx.x * 16;

    {
        const int base = tid * 8;
        const int r = base >> 7, c = base & (CC - 1);
        const float4* px =
            reinterpret_cast<const float4*>(x + (size_t)r0 * CC + base);
        const float4 a0 = px[0], a1 = px[1];
        xs[r][c + 0] = (__bf16)a0.x; xs[r][c + 1] = (__bf16)a0.y;
        xs[r][c + 2] = (__bf16)a0.z; xs[r][c + 3] = (__bf16)a0.w;
        xs[r][c + 4] = (__bf16)a1.x; xs[r][c + 5] = (__bf16)a1.y;
        xs[r][c + 6] = (__bf16)a1.z; xs[r][c + 7] = (__bf16)a1.w;
    }
    __syncthreads();

    const int w    = tid >> 5;
    const int lane = tid & 31;

    // Each wave covers 4 column tiles: cols [w*64, w*64+64).
    v8f acc[4] = {};
#pragma unroll
    for (int kc = 0; kc < 4; ++kc) {
        const v16bf a = frag_a_lds(&xs[0][0], lane, kc * 32);
#pragma unroll
        for (int t = 0; t < 4; ++t) {
            const v16bf b = frag_b_packed(Wfpb, w * 4 + t, kc, lane);
            acc[t] = __builtin_amdgcn_wmma_f32_16x16x32_bf16(
                false, a, false, b, (short)0, acc[t], false, false);
        }
    }

    const int n  = lane & 15;
    const int mb = (lane >> 4) << 3;
#pragma unroll
    for (int t = 0; t < 4; ++t)
#pragma unroll
        for (int e = 0; e < 8; ++e)
            ls[mb + e][w * 64 + t * 16 + n] = acc[t][e];
    __syncthreads();

    // Softmax + scatter: wave w handles rows 2w and 2w+1.
#pragma unroll
    for (int rr = 0; rr < 2; ++rr) {
        const int r = w * 2 + rr;
        float m = -3.402823466e38f;
        for (int c = lane; c < FPD; c += 32) m = fmaxf(m, ls[r][c]);
#pragma unroll
        for (int o = 16; o > 0; o >>= 1) m = fmaxf(m, __shfl_xor(m, o));
        float s = 0.f;
        for (int c = lane; c < FPD; c += 32) {
            const float e = __expf(ls[r][c] - m);
            ls[r][c] = e;
            s += e;
        }
#pragma unroll
        for (int o = 16; o > 0; o >>= 1) s += __shfl_xor(s, o);
        const float inv = 1.0f / s;
        const int g = batch[r0 + r];
        float* po = out + (size_t)g * FPD;
        for (int c = lane; c < FPD; c += 32) atomicAdd(po + c, ls[r][c] * inv);
    }
}

// ---------------------------------------------------------------------------
// Launch: pack weights (bf16, fragment-major) + zero fingerprint; per layer:
// neigh = 0 -> edge scatter -> gemm+tanh (in-place x after layer 0) ->
// fingerprint GEMM/softmax/scatter.
// Workspace: x (51.2 MB) + neigh (51.2 MB) + packed weights (~0.6 MB).
// ---------------------------------------------------------------------------
extern "C" void kernel_launch(void* const* d_in, const int* in_sizes, int n_in,
                              void* d_out, int out_size, void* d_ws, size_t ws_size,
                              hipStream_t stream) {
    const float* x_in = (const float*)d_in[0];
    const float* Ws   = (const float*)d_in[1];
    const float* bs   = (const float*)d_in[2];
    const float* Wn   = (const float*)d_in[3];
    const float* bn   = (const float*)d_in[4];
    const float* Wfp  = (const float*)d_in[5];
    const int*   ei   = (const int*)d_in[6];
    const int*   bat  = (const int*)d_in[7];
    float*       out  = (float*)d_out;

    float*  x_cur = (float*)d_ws;
    float*  neigh = x_cur + (size_t)NN * CC;
    __bf16* Wsb   = (__bf16*)(neigh + (size_t)NN * CC);   // 32-B aligned
    __bf16* Wnb   = Wsb + (size_t)LL * WSB_PER_LAYER;
    __bf16* Wfpb  = Wnb + (size_t)LL * WSB_PER_LAYER;

    // One-time weight repack to fragment-major bf16 (layers contiguous).
    {
        const int fr_ws  = LL * (CC / 16) * 4 * 32;    // 3072 fragments
        const int fr_wfp = LL * (FPD / 16) * 4 * 32;   // 12288 fragments
        pack_weights_kernel<<<(fr_ws + 255) / 256, 256, 0, stream>>>(Ws, Wsb, fr_ws);
        pack_weights_kernel<<<(fr_ws + 255) / 256, 256, 0, stream>>>(Wn, Wnb, fr_ws);
        pack_weights_kernel<<<(fr_wfp + 255) / 256, 256, 0, stream>>>(Wfp, Wfpb, fr_wfp);
    }

    // fingerprint = 0  (G*FP = 524288 f32 = 131072 float4)
    zero_f32_kernel<<<512, 256, 0, stream>>>(out, (GG * FPD) / 4);

    for (int layer = 0; layer < LL; ++layer) {
        const float* xsrc = (layer == 0) ? x_in : x_cur;

        // neigh = 0  (N*C = 12.8M f32 = 3.2M float4)
        zero_f32_kernel<<<12500, 256, 0, stream>>>(neigh, (NN * CC) / 4);

        // neigh[row] += x[col]   (one wave per edge, 8 edges per block)
        gather_scatter_kernel<<<EE / 8, 256, 0, stream>>>(xsrc, ei, neigh);

        // x = tanh(x@Ws^T + bs + neigh@Wn^T + bn)
        layer_gemm_tanh_kernel<<<NN / 16, 256, 0, stream>>>(
            xsrc, neigh,
            Wsb + (size_t)layer * WSB_PER_LAYER, bs + (size_t)layer * CC,
            Wnb + (size_t)layer * WSB_PER_LAYER, bn + (size_t)layer * CC,
            x_cur);

        // fingerprint += segment_sum(softmax(x@Wfp^T), batch)
        fp_softmax_scatter_kernel<<<NN / 16, 256, 0, stream>>>(
            x_cur, Wfpb + (size_t)layer * WFPB_PER_LAYER, bat, out);
    }
}
